// ModulatedConv2D_90967407330081
// MI455X (gfx1250) — matrix operand order repaired and checked by reading, not compiled, not run on hardware
//
#include <hip/hip_runtime.h>

// ---- problem constants --------------------------------------------------
#define BB    8
#define CIN   32
#define COUT  32
#define HH    512
#define WW    512
#define LAT   512
#define TAPS  9            // 3x3
#define TPX   256          // pixels per block (along W)
#define XT_COLS 260        // 258 data cols (halo) + pad
#define OPAD  260          // padded pixel stride in output LDS tile

typedef __attribute__((ext_vector_type(16))) __bf16 v16bf;
typedef __attribute__((ext_vector_type(8)))  __bf16 v8bf;
typedef __attribute__((ext_vector_type(8)))  float  v8f;

__device__ __forceinline__ unsigned short f2bf(float f) {
    unsigned int u = __float_as_uint(f);
    unsigned int r = u + 0x7fffu + ((u >> 16) & 1u);   // round-to-nearest-even
    return (unsigned short)(r >> 16);
}

__device__ __forceinline__ v16bf ldfrag(const unsigned short* p, int second_off) {
    v8bf lo = *reinterpret_cast<const v8bf*>(p);
    v8bf hi = *reinterpret_cast<const v8bf*>(p + second_off);
    return __builtin_shufflevector(lo, hi, 0,1,2,3,4,5,6,7,8,9,10,11,12,13,14,15);
}

// ---- prologue: style modulation + demod, emit bf16 weights --------------
// wsB layout: [b][tap][cout][cin]  (cin contiguous -> matches WMMA B frag)
__global__ void __launch_bounds__(256)
modw_kernel(const float* __restrict__ y, const float* __restrict__ w,
            const float* __restrict__ w_dense, const float* __restrict__ b_dense,
            unsigned short* __restrict__ wsB) {
    __shared__ float s_sm[BB * CIN];
    __shared__ float d_sm[BB * COUT];
    const int tid = threadIdx.x;
    const float dense_coef = 0.04419417382415922f;   // 1/sqrt(512)
    const float rc         = 0.05892556509887896f;   // 1/sqrt(3*3*32)

    // s[b][cin] = y[b] . (w_dense * coef)[:,cin] + b_dense + 1
    {
        const int b = tid >> 5, cin = tid & 31;
        float acc = 0.f;
        for (int l = 0; l < LAT; ++l)
            acc += y[b * LAT + l] * w_dense[l * CIN + cin];
        s_sm[tid] = acc * dense_coef + b_dense[cin] + 1.0f;
    }
    __syncthreads();

    // d[b][cout] = rsqrt(sum_{tap,cin} (rc*w*s)^2 + 1e-8)
    {
        const int b = tid >> 5, cout = tid & 31;
        float acc = 0.f;
        for (int t = 0; t < TAPS; ++t)
            for (int cin = 0; cin < CIN; ++cin) {
                float wv = rc * w[(t * CIN + cin) * COUT + cout] * s_sm[b * CIN + cin];
                acc += wv * wv;
            }
        d_sm[tid] = rsqrtf(acc + 1e-8f);
    }
    __syncthreads();

    // emit bf16 modulated+demodulated weights
    for (int i = tid; i < BB * TAPS * COUT * CIN; i += 256) {
        const int cin  = i & 31;
        const int cout = (i >> 5) & 31;
        const int t    = (i >> 10) % TAPS;
        const int b    = i / (TAPS * COUT * CIN);
        float val = rc * w[(t * CIN + cin) * COUT + cout]
                       * s_sm[b * CIN + cin] * d_sm[b * COUT + cout];
        wsB[i] = f2bf(val);
    }
}

// ---- main conv: implicit GEMM via v_wmma_f32_16x16x32_bf16 --------------
// Block: 256 px x 32 cout for one (b, row). 8 waves; each wave owns two
// 16-pixel M-tiles sharing B fragments -> 36 WMMAs/wave-group, ds:wmma 2:1.
__global__ void __launch_bounds__(256)
modconv_kernel(const float* __restrict__ x, const unsigned short* __restrict__ wsB,
               float* __restrict__ out) {
    __shared__ __attribute__((aligned(16))) unsigned short xs[3 * XT_COLS * CIN]; // 49920 B
    __shared__ __attribute__((aligned(16))) unsigned short wl[TAPS * COUT * CIN]; // 18432 B
    __shared__ __attribute__((aligned(16))) float         olds[COUT * OPAD];      // 33280 B

    const int tid  = threadIdx.x;
    const int gid  = blockIdx.x;                    // B * H * (W/TPX)
    const int xt   = gid & 1;                       // W/256 = 2
    const int yrow = (gid >> 1) & (HH - 1);
    const int b    = gid >> 10;                     // /(2*512)
    const int x0   = xt * TPX;

    // ---- stage input tile, interior: rows yrow-1..+1, cols x0..x0+255
    // (pure shift/mask decode; interior x is always in range)
    for (int i = tid; i < 3 * CIN * TPX; i += 256) {
        const int c   = (i & 255) + 1;              // local col 1..256
        const int cin = (i >> 8) & 31;
        const int r   = i >> 13;
        const int yg  = yrow + r - 1;
        float v = 0.f;
        if (yg >= 0 && yg < HH)
            v = x[(((b * CIN + cin) * HH) + yg) * WW + x0 + c - 1];
        xs[(r * XT_COLS + c) * CIN + cin] = f2bf(v);
    }
    // ---- halo columns (local col 0 and 257): 3*32*2 = 192 entries
    if (tid < 192) {
        const int r   = tid >> 6;
        const int rem = tid & 63;
        const int cin = rem >> 1;
        const int c   = (rem & 1) ? 257 : 0;
        const int yg  = yrow + r - 1;
        const int xg  = x0 + c - 1;
        float v = 0.f;
        if (yg >= 0 && yg < HH && xg >= 0 && xg < WW)
            v = x[(((b * CIN + cin) * HH) + yg) * WW + xg];
        xs[(r * XT_COLS + c) * CIN + cin] = f2bf(v);
    }
    // ---- stage this batch's weights (as dwords)
    {
        const unsigned int* src = (const unsigned int*)(wsB + b * TAPS * COUT * CIN);
        unsigned int*       dst = (unsigned int*)wl;
        for (int i = tid; i < TAPS * COUT * CIN / 2; i += 256) dst[i] = src[i];
    }
    __syncthreads();

    // ---- per-wave WMMA: 32 pixels x 32 couts, K = 9 taps x 32 cin
    const int lane  = tid & 31;
    const int wave  = tid >> 5;
    const int m_off = wave * 32;        // pixel tile base within 256
    const int half  = lane >> 4;        // K-half selector
    const int l15   = lane & 15;

    v8f c00 = {0.f,0.f,0.f,0.f,0.f,0.f,0.f,0.f};
    v8f c01 = {0.f,0.f,0.f,0.f,0.f,0.f,0.f,0.f};
    v8f c10 = {0.f,0.f,0.f,0.f,0.f,0.f,0.f,0.f};
    v8f c11 = {0.f,0.f,0.f,0.f,0.f,0.f,0.f,0.f};

#pragma unroll
    for (int t = 0; t < TAPS; ++t) {
        const int dy = t / 3, dx = t % 3;            // constant-folded (unrolled)
        // A fragments: lane half h holds cin [8h..8h+7] and [16+8h..23+8h]
        const int p = m_off + l15 + dx;              // local col
        const unsigned short* arow = &xs[(dy * XT_COLS + p) * CIN + 8 * half];
        v16bf a0 = ldfrag(arow, 16);
        v16bf a1 = ldfrag(arow + 16 * CIN, 16);      // second M-tile (+16 px)

        // B fragments: lane n holds cin [8h..8h+7] and [16+8h..23+8h] for cout=n
        v16bf bm0 = ldfrag(&wl[(t * COUT + l15) * CIN + 8 * half], 16);
        v16bf bm1 = ldfrag(&wl[(t * COUT + 16 + l15) * CIN + 8 * half], 16);

        c00 = __builtin_amdgcn_wmma_f32_16x16x32_bf16(false, a0, false, bm0,
                                                      (short)0, c00, false, false);
        c01 = __builtin_amdgcn_wmma_f32_16x16x32_bf16(false, a0, false, bm1,
                                                      (short)0, c01, false, false);
        c10 = __builtin_amdgcn_wmma_f32_16x16x32_bf16(false, a1, false, bm0,
                                                      (short)0, c10, false, false);
        c11 = __builtin_amdgcn_wmma_f32_16x16x32_bf16(false, a1, false, bm1,
                                                      (short)0, c11, false, false);
    }

    // ---- scatter C into padded LDS tile: olds[cout][pixel]
#pragma unroll
    for (int v = 0; v < 8; ++v) {
        const int pix = m_off + v + 8 * half;        // C: VGPR v -> M=v (+8 hi lanes)
        olds[l15 * OPAD + pix]             = c00[v];
        olds[(16 + l15) * OPAD + pix]      = c01[v];
        olds[l15 * OPAD + pix + 16]        = c10[v];
        olds[(16 + l15) * OPAD + pix + 16] = c11[v];
    }
    __syncthreads();

    // ---- coalesced float4 stores to out[b][cout][yrow][x0..x0+255]
    for (int j = tid; j < COUT * (TPX / 4); j += 256) {
        const int cout = j >> 6;
        const int p4   = j & 63;
        const float4 val = *reinterpret_cast<const float4*>(&olds[cout * OPAD + p4 * 4]);
        *reinterpret_cast<float4*>(
            &out[(((b * COUT + cout) * HH) + yrow) * WW + x0 + p4 * 4]) = val;
    }
}

extern "C" void kernel_launch(void* const* d_in, const int* in_sizes, int n_in,
                              void* d_out, int out_size, void* d_ws, size_t ws_size,
                              hipStream_t stream) {
    const float* x       = (const float*)d_in[0];
    const float* y       = (const float*)d_in[1];
    const float* w       = (const float*)d_in[2];
    const float* w_dense = (const float*)d_in[3];
    const float* b_dense = (const float*)d_in[4];
    float* out = (float*)d_out;
    unsigned short* wsB = (unsigned short*)d_ws;   // B*9*32*32 bf16 = 147456 B

    modw_kernel<<<1, 256, 0, stream>>>(y, w, w_dense, b_dense, wsB);
    const int nblk = BB * HH * (WW / TPX);         // 8192
    modconv_kernel<<<nblk, 256, 0, stream>>>(x, wsB, out);
}
